// AttentionModel_39848706572778
// MI455X (gfx1250) — matrix operand order repaired and checked by reading, not compiled, not run on hardware
//
#include <hip/hip_runtime.h>

#define WG 256

typedef __bf16 bf16_t;
typedef bf16_t v16bf __attribute__((ext_vector_type(16)));
typedef float  v8f   __attribute__((ext_vector_type(8)));

struct alignas(16) U128 { unsigned int w[4]; };
union Frag { v16bf v; U128 q[2]; };
struct alignas(8) US4 { unsigned short s[4]; };

__device__ __forceinline__ unsigned short f2bf(float f) {
  unsigned int u = __float_as_uint(f);
  u += 0x7FFFu + ((u >> 16) & 1u);   // round-to-nearest-even
  return (unsigned short)(u >> 16);
}

// ---------------- f32 -> bf16 convert (vectorized x4) ----------------
__global__ __launch_bounds__(WG) void k_f32_to_bf16(
    const float* __restrict__ in, unsigned short* __restrict__ out, int n4) {
  int i = blockIdx.x * WG + threadIdx.x;
  if (i >= n4) return;
  float4 f = reinterpret_cast<const float4*>(in)[i];
  US4 o;
  o.s[0] = f2bf(f.x); o.s[1] = f2bf(f.y); o.s[2] = f2bf(f.z); o.s[3] = f2bf(f.w);
  reinterpret_cast<US4*>(out)[i] = o;
}

// ---------------- bf16 tile transpose (per batch) ----------------
__global__ __launch_bounds__(WG) void k_transpose_bf16(
    const unsigned short* __restrict__ in, unsigned short* __restrict__ out,
    int rows, int cols) {
  __shared__ unsigned short tile[32][33];
  int bz = blockIdx.z;
  const unsigned short* ib = in + (size_t)bz * rows * cols;
  unsigned short*       ob = out + (size_t)bz * rows * cols;
  int c0 = blockIdx.x * 32;
  int r0 = blockIdx.y * 32;
  int tx = threadIdx.x & 31;
  int ty = threadIdx.x >> 5;            // 0..7
#pragma unroll
  for (int i = 0; i < 32; i += 8)
    tile[ty + i][tx] = ib[(size_t)(r0 + ty + i) * cols + (c0 + tx)];
  __syncthreads();
#pragma unroll
  for (int i = 0; i < 32; i += 8)
    ob[(size_t)(c0 + ty + i) * rows + (r0 + tx)] = tile[tx][ty + i];
}

// ---------------- row softmax: f32 scores -> bf16 attn ----------------
__global__ __launch_bounds__(WG) void k_softmax(
    const float* __restrict__ scores, unsigned short* __restrict__ attn, int S) {
  __shared__ float red[WG];
  const float* row = scores + (size_t)blockIdx.x * S;
  unsigned short* orow = attn + (size_t)blockIdx.x * S;
  int t = threadIdx.x;
  float vals[8];
  float m = -3.4e38f;
#pragma unroll
  for (int j = 0; j < 8; ++j) { float v = row[t + j * WG]; vals[j] = v; m = fmaxf(m, v); }
  red[t] = m; __syncthreads();
  for (int s = WG / 2; s > 0; s >>= 1) { if (t < s) red[t] = fmaxf(red[t], red[t + s]); __syncthreads(); }
  m = red[0]; __syncthreads();
  float sum = 0.f;
#pragma unroll
  for (int j = 0; j < 8; ++j) { vals[j] = __expf(vals[j] - m); sum += vals[j]; }
  red[t] = sum; __syncthreads();
  for (int s = WG / 2; s > 0; s >>= 1) { if (t < s) red[t] += red[t + s]; __syncthreads(); }
  float inv = 1.0f / red[0];
#pragma unroll
  for (int j = 0; j < 8; ++j) orow[t + j * WG] = f2bf(vals[j] * inv);
}

// ---------------- NT GEMM: C[m,n] = sum_k A[m,k]*B[n,k], bf16 in, f32 acc ----
// Block: 256 thr = 8 waves, each wave 32x64 (2x4 tiles of 16x16), block 128x128.
enum { EPI_BIAS_BF16 = 0, EPI_SCALE_MASK_F32 = 1, EPI_F32 = 2 };

template <int EPI>
__global__ __launch_bounds__(WG) void k_gemm_nt_bf16(
    const unsigned short* __restrict__ A, int lda, long long strideA,
    const unsigned short* __restrict__ B, int ldb, long long strideB,
    const float* __restrict__ bias,
    const float* __restrict__ mask, long long strideMask, float scale,
    unsigned short* __restrict__ outBf,
    float* __restrict__ outF, int ldo, long long strideO,
    int K) {
  const int lane = threadIdx.x & 31;
  const int wave = threadIdx.x >> 5;
  const int wm = wave & 3;
  const int wn = wave >> 2;
  const int bz = blockIdx.z;
  const int m0 = blockIdx.y * 128 + wm * 32;
  const int n0 = blockIdx.x * 128 + wn * 64;

  const unsigned short* Ab = A + (size_t)bz * strideA;
  const unsigned short* Bb = B + (size_t)bz * strideB;

  const int r  = lane & 15;
  const int kA = (lane >> 4) * 8;    // A frag (16x32, 16-bit): runs [kA..kA+7], [kA+16..kA+23]
  const int kB = (lane >> 4) * 8;    // B frag (32x16, 16-bit): runs [kB..kB+7], [kB+16..kB+23]

  const unsigned short* ap[2];
  const unsigned short* bp[4];
#pragma unroll
  for (int t = 0; t < 2; ++t) ap[t] = Ab + (size_t)(m0 + t * 16 + r) * lda + kA;
#pragma unroll
  for (int u = 0; u < 4; ++u) bp[u] = Bb + (size_t)(n0 + u * 16 + r) * ldb + kB;

  v8f acc[2][4];
#pragma unroll
  for (int t = 0; t < 2; ++t)
#pragma unroll
    for (int u = 0; u < 4; ++u)
      acc[t][u] = (v8f){0.f, 0.f, 0.f, 0.f, 0.f, 0.f, 0.f, 0.f};

  for (int k0 = 0; k0 < K; k0 += 32) {
    Frag af[2], bfr[4];
#pragma unroll
    for (int t = 0; t < 2; ++t) {
      af[t].q[0] = *reinterpret_cast<const U128*>(ap[t] + k0);
      af[t].q[1] = *reinterpret_cast<const U128*>(ap[t] + k0 + 16);
    }
#pragma unroll
    for (int u = 0; u < 4; ++u) {
      bfr[u].q[0] = *reinterpret_cast<const U128*>(bp[u] + k0);
      bfr[u].q[1] = *reinterpret_cast<const U128*>(bp[u] + k0 + 16);
    }
#pragma unroll
    for (int t = 0; t < 2; ++t)
#pragma unroll
      for (int u = 0; u < 4; ++u)
        acc[t][u] = __builtin_amdgcn_wmma_f32_16x16x32_bf16(
            false, af[t].v, false, bfr[u].v, (short)0, acc[t][u], false, false);
  }

  const int mo = (lane >> 4) * 8;
  const int nn = lane & 15;
#pragma unroll
  for (int t = 0; t < 2; ++t)
#pragma unroll
    for (int u = 0; u < 4; ++u)
#pragma unroll
      for (int e = 0; e < 8; ++e) {
        int gm = m0 + t * 16 + mo + e;
        int gn = n0 + u * 16 + nn;
        float v = acc[t][u][e];
        if (EPI == EPI_BIAS_BF16) {
          v += bias[gn];
          outBf[(size_t)gm * ldo + gn] = f2bf(v);
        } else if (EPI == EPI_SCALE_MASK_F32) {
          v = v * scale + mask[(size_t)bz * strideMask + (size_t)gm * ldo + gn];
          outF[(size_t)bz * strideO + (size_t)gm * ldo + gn] = v;
        } else {
          outF[(size_t)bz * strideO + (size_t)gm * ldo + gn] = v;
        }
      }
}

extern "C" void kernel_launch(void* const* d_in, const int* in_sizes, int n_in,
                              void* d_out, int out_size, void* d_ws, size_t ws_size,
                              hipStream_t stream) {
  const int Bn = 4, S = 2048, E = 1024, BS = Bn * S;
  const float inv_scale = 1.0f / 32.0f;  // 1/sqrt(1024)

  const float* q_in = (const float*)d_in[0];
  const float* k_in = (const float*)d_in[1];
  const float* v_in = (const float*)d_in[2];
  const float* mask = (const float*)d_in[3];
  const float* Wq   = (const float*)d_in[4];
  const float* bq   = (const float*)d_in[5];
  const float* Wk   = (const float*)d_in[6];
  const float* bk   = (const float*)d_in[7];
  const float* Wv   = (const float*)d_in[8];
  const float* bv   = (const float*)d_in[9];

  char* ws = (char*)d_ws;
  const size_t MB = 1024 * 1024;
  // Workspace map (166 MB total, with aliasing of freed regions):
  // [0,48)   bf16 inputs X_q/X_k/X_v   (freed after projections)
  // [48,54)  bf16 weights Wq/Wk/Wv
  // [54,102) bf16 q / k / v projections
  // [102,166) f32 scores (B,S,S)
  // alias: attn bf16 at [0,32), vT bf16 at [32,48)
  unsigned short* bfX[3] = {(unsigned short*)(ws + 0 * MB),
                            (unsigned short*)(ws + 16 * MB),
                            (unsigned short*)(ws + 32 * MB)};
  unsigned short* bfW[3] = {(unsigned short*)(ws + 48 * MB),
                            (unsigned short*)(ws + 50 * MB),
                            (unsigned short*)(ws + 52 * MB)};
  unsigned short* qb = (unsigned short*)(ws + 54 * MB);
  unsigned short* kb = (unsigned short*)(ws + 70 * MB);
  unsigned short* vb = (unsigned short*)(ws + 86 * MB);
  float* scores      = (float*)(ws + 102 * MB);
  unsigned short* attn = (unsigned short*)(ws + 0 * MB);   // reuse bfX[0..1]
  unsigned short* vT   = (unsigned short*)(ws + 32 * MB);  // reuse bfX[2]

  // 1) convert inputs + weights to bf16
  int nX4 = BS * E / 4, nW4 = E * E / 4;
  k_f32_to_bf16<<<(nX4 + WG - 1) / WG, WG, 0, stream>>>(q_in, bfX[0], nX4);
  k_f32_to_bf16<<<(nX4 + WG - 1) / WG, WG, 0, stream>>>(k_in, bfX[1], nX4);
  k_f32_to_bf16<<<(nX4 + WG - 1) / WG, WG, 0, stream>>>(v_in, bfX[2], nX4);
  k_f32_to_bf16<<<(nW4 + WG - 1) / WG, WG, 0, stream>>>(Wq, bfW[0], nW4);
  k_f32_to_bf16<<<(nW4 + WG - 1) / WG, WG, 0, stream>>>(Wk, bfW[1], nW4);
  k_f32_to_bf16<<<(nW4 + WG - 1) / WG, WG, 0, stream>>>(Wv, bfW[2], nW4);

  // 2) projections: q/k/v = X @ W^T + b   (M=8192, N=1024, K=1024, NT)
  dim3 gp(E / 128, BS / 128, 1);
  k_gemm_nt_bf16<EPI_BIAS_BF16><<<gp, WG, 0, stream>>>(
      bfX[0], E, 0, bfW[0], E, 0, bq, nullptr, 0, 0.f, qb, nullptr, E, 0, E);
  k_gemm_nt_bf16<EPI_BIAS_BF16><<<gp, WG, 0, stream>>>(
      bfX[1], E, 0, bfW[1], E, 0, bk, nullptr, 0, 0.f, kb, nullptr, E, 0, E);
  k_gemm_nt_bf16<EPI_BIAS_BF16><<<gp, WG, 0, stream>>>(
      bfX[2], E, 0, bfW[2], E, 0, bv, nullptr, 0, 0.f, vb, nullptr, E, 0, E);

  // 3) transpose v (per batch S x E -> E x S) so PV GEMM is NT
  dim3 gt(E / 32, S / 32, Bn);
  k_transpose_bf16<<<gt, WG, 0, stream>>>(vb, vT, S, E);

  // 4) scores = (q @ k^T) * inv_scale + mask   (per batch, M=N=2048, K=1024)
  dim3 gs(S / 128, S / 128, Bn);
  k_gemm_nt_bf16<EPI_SCALE_MASK_F32><<<gs, WG, 0, stream>>>(
      qb, E, (long long)S * E, kb, E, (long long)S * E,
      nullptr, mask, (long long)S * S, inv_scale,
      nullptr, scores, S, (long long)S * S, E);

  // 5) softmax rows -> bf16 attn
  k_softmax<<<Bn * S, WG, 0, stream>>>(scores, attn, S);

  // 6) out = attn @ v  (via vT, NT: M=2048, N=1024, K=2048 per batch)
  dim3 go(E / 128, S / 128, Bn);
  k_gemm_nt_bf16<EPI_F32><<<go, WG, 0, stream>>>(
      attn, S, (long long)S * S, vT, S, (long long)S * E,
      nullptr, nullptr, 0, 0.f,
      nullptr, (float*)d_out, E, (long long)S * E, S);
}
